// MambaPlusPlus_layer_53936199303397
// MI455X (gfx1250) — matrix operand order repaired and checked by reading, not compiled, not run on hardware
//
#include <hip/hip_runtime.h>
#include <cstdint>

typedef __attribute__((ext_vector_type(16))) _Float16 v16h;
typedef __attribute__((ext_vector_type(8)))  float    v8f;
typedef __attribute__((ext_vector_type(4)))  float    f32x4;

union FragH { v16h h; f32x4 q[2]; };

#define M_TOT  32768   // B*L tokens
#define K_E    1024    // embed dim
#define N_PAD  2176    // 17 * 128
#define N_REAL 2112    // 1024 (a) + 1024 (b) + 64 (dx)
#define NCH    8192    // B*H*D scan channels
#define CHUNK  256     // scan chunk length
#define NCHK   16      // 4096 / CHUNK chunks per sequence

// ---------------------------------------------------------------------------
// Weight packing: Wcat_t[n][k] (f16, transposed for WMMA B-fragments)
// n in [0,1024): Wa[h=n/64][k][n%64]; [1024,2048): Wb; [2048,2112): Wd; pad 0
// ---------------------------------------------------------------------------
__global__ void k_pack_wcat(const float* __restrict__ Wa, const float* __restrict__ Wb,
                            const float* __restrict__ Wd, _Float16* __restrict__ Wt)
{
  int i = blockIdx.x * 256 + threadIdx.x;
  if (i >= N_PAD * K_E) return;
  int n = i >> 10, k = i & 1023;
  float v = 0.f;
  if (n < 1024)        v = Wa[(size_t)(n >> 6) * 65536 + (size_t)k * 64 + (n & 63)];
  else if (n < 2048) { int nn = n - 1024;
                       v = Wb[(size_t)(nn >> 6) * 65536 + (size_t)k * 64 + (nn & 63)]; }
  else if (n < N_REAL) v = Wd[(size_t)k * 64 + (n - 2048)];
  Wt[(size_t)n * K_E + k] = (_Float16)v;
}

// dst[n*K + k] = (f16) src[k*N + n]   (row-major [K][N] -> transposed f16 [N][K])
__global__ void k_pack_trans(const float* __restrict__ src, _Float16* __restrict__ dst,
                             int K, int N)
{
  int i = blockIdx.x * blockDim.x + threadIdx.x;
  if (i >= K * N) return;
  int n = i / K, k = i % K;
  dst[i] = (_Float16)src[(size_t)k * N + n];
}

// ---------------------------------------------------------------------------
// Fused projection GEMM: P[32768, 2112] = emb @ [Wa|Wb|Wd], epilogue:
//   n<1024  : a  = tanh(p + ba)   -> Ag
//   n<2048  : b  =       p + bb   -> Bg
//   n<2112  : dx =       p + bd   -> DXg
// Block 128x128, K-step 32, 8 waves (wave tile 32x64 = 8 accum tiles).
// B tile: double-buffered CDNA5 async-to-LDS DMA pipelined one K-step ahead
// (in-order ASYNCcnt completion => s_wait_asynccnt 2 retires current buffer).
// ---------------------------------------------------------------------------
__global__ __launch_bounds__(256) void k_gemm_proj(
    const float* __restrict__ X, const _Float16* __restrict__ Wt,
    const float* __restrict__ ba, const float* __restrict__ bb, const float* __restrict__ bd,
    float* __restrict__ Ag, float* __restrict__ Bg, float* __restrict__ DXg)
{
  __shared__ __align__(16) _Float16 As[2][128][32];   // [buf][m][k]
  __shared__ __align__(16) _Float16 Bs[2][128][32];   // [buf][n][k] (pre-transposed)
  const int tid = threadIdx.x;
  const int wid = tid >> 5, lane = tid & 31;
  const int r = lane & 15, hf = lane >> 4;
  const int wm = wid & 3, wn = wid >> 2;
  const int m0 = blockIdx.y * 128, n0 = blockIdx.x * 128;

  auto issueB = [&](int kb, int buf) {
    #pragma unroll
    for (int i = 0; i < 2; ++i) {
      int c = tid + i * 256, n = c >> 2, kc = c & 3;
      uint32_t ldsoff = (uint32_t)(uintptr_t)(&Bs[buf][n][kc * 8]);
      uint64_t gaddr  = (uint64_t)(uintptr_t)(Wt + (size_t)(n0 + n) * K_E + kb * 32 + kc * 8);
      asm volatile("global_load_async_to_lds_b128 %0, %1, off"
                   :: "v"(ldsoff), "v"(gaddr) : "memory");
    }
  };

  const v8f z0 = {0.f,0.f,0.f,0.f,0.f,0.f,0.f,0.f};
  v8f acc[2][4];
  #pragma unroll
  for (int mt = 0; mt < 2; ++mt)
    #pragma unroll
    for (int nt = 0; nt < 4; ++nt) acc[mt][nt] = z0;

  issueB(0, 0);                                   // prime the pipeline

  for (int kb = 0; kb < 32; ++kb) {
    const int cur = kb & 1;
    // stage A (f32 -> f16, needs conversion so it goes through VGPRs): 128x32
    #pragma unroll
    for (int i = 0; i < 4; ++i) {
      int c = tid + i * 256, m = c >> 3, kc = c & 7;
      f32x4 v = *(const f32x4*)(X + (size_t)(m0 + m) * K_E + kb * 32 + kc * 4);
      _Float16* d = &As[cur][m][kc * 4];
      d[0] = (_Float16)v.x; d[1] = (_Float16)v.y; d[2] = (_Float16)v.z; d[3] = (_Float16)v.w;
    }
    // prefetch next B tile into the other buffer; retire current buffer's DMA
    if (kb + 1 < 32) {
      issueB(kb + 1, cur ^ 1);
      asm volatile("s_wait_asynccnt 2" ::: "memory");
    } else {
      asm volatile("s_wait_asynccnt 0" ::: "memory");
    }
    __syncthreads();

    FragH af[2], bf[4];
    #pragma unroll
    for (int mt = 0; mt < 2; ++mt) {
      const _Float16* rp = &As[cur][wm * 32 + mt * 16 + r][0];
      af[mt].q[0] = *(const f32x4*)(rp + hf * 8);
      af[mt].q[1] = *(const f32x4*)(rp + 16 + hf * 8);
    }
    #pragma unroll
    for (int nt = 0; nt < 4; ++nt) {
      const _Float16* rp = &Bs[cur][wn * 64 + nt * 16 + r][0];
      bf[nt].q[0] = *(const f32x4*)(rp + hf * 8);
      bf[nt].q[1] = *(const f32x4*)(rp + 16 + hf * 8);
    }
    #pragma unroll
    for (int mt = 0; mt < 2; ++mt)
      #pragma unroll
      for (int nt = 0; nt < 4; ++nt)
        acc[mt][nt] = __builtin_amdgcn_wmma_f32_16x16x32_f16(
            false, af[mt].h, false, bf[nt].h, (short)0, acc[mt][nt], false, false);
    __syncthreads();
  }

  // epilogue: segment choice is wave-uniform per 16-col tile
  // (1024 / 2048 / 2112 are all multiples of 16, tile bases are 16-aligned)
  #pragma unroll
  for (int mt = 0; mt < 2; ++mt)
    #pragma unroll
    for (int nt = 0; nt < 4; ++nt) {
      const int tcol = n0 + wn * 64 + nt * 16;   // uniform across the wave
      if (tcol >= N_REAL) continue;
      const int col = tcol + r;
      if (tcol < 1024) {
        const float bias = ba[col];
        #pragma unroll
        for (int v = 0; v < 8; ++v) {
          int row = m0 + wm * 32 + mt * 16 + v + 8 * hf;
          Ag[(size_t)row * 1024 + col] = tanhf(acc[mt][nt][v] + bias);
        }
      } else if (tcol < 2048) {
        const float bias = bb[col - 1024];
        #pragma unroll
        for (int v = 0; v < 8; ++v) {
          int row = m0 + wm * 32 + mt * 16 + v + 8 * hf;
          Bg[(size_t)row * 1024 + (col - 1024)] = acc[mt][nt][v] + bias;
        }
      } else {
        const float bias = bd[col - 2048];
        #pragma unroll
        for (int v = 0; v < 8; ++v) {
          int row = m0 + wm * 32 + mt * 16 + v + 8 * hf;
          DXg[(size_t)row * 64 + (col - 2048)] = acc[mt][nt][v] + bias;
        }
      }
    }
}

// ---------------------------------------------------------------------------
// Chunked associative scan of h_t = a_t*h_{t-1} + b_t.
// (A1,B1)∘(A2,B2) = (A1*A2, A2*B1+B2);  h_out = A*h_in + B.
// ---------------------------------------------------------------------------
__global__ __launch_bounds__(256) void k_scan_p1(
    const float* __restrict__ Ag, const float* __restrict__ Bg,
    float* __restrict__ SA, float* __restrict__ SB)
{
  int g  = blockIdx.x * 256 + threadIdx.x;   // NCH * NCHK
  int ch = g & (NCH - 1), c = g >> 13;
  int b = ch >> 10, hd = ch & 1023;
  size_t base = (size_t)b * 4096 * 1024 + (size_t)c * CHUNK * 1024 + hd;
  float A = 1.f, Bv = 0.f;
  for (int t = 0; t < CHUNK; ++t) {
    size_t idx = base + (size_t)t * 1024;
    if (t + 4 < CHUNK) {
      __builtin_prefetch(Ag + idx + 4 * 1024, 0, 1);
      __builtin_prefetch(Bg + idx + 4 * 1024, 0, 1);
    }
    float a = Ag[idx], bb_ = Bg[idx];
    A *= a;
    Bv = a * Bv + bb_;
  }
  SA[(size_t)c * NCH + ch] = A;
  SB[(size_t)c * NCH + ch] = Bv;
}

__global__ __launch_bounds__(256) void k_scan_p2(
    const float* __restrict__ SA, const float* __restrict__ SB,
    float* __restrict__ SC)
{
  int ch = blockIdx.x * 256 + threadIdx.x;   // NCH
  float cur = 0.f;
  #pragma unroll
  for (int c = 0; c < NCHK; ++c) {
    SC[(size_t)c * NCH + ch] = cur;          // carry entering chunk c
    cur = SA[(size_t)c * NCH + ch] * cur + SB[(size_t)c * NCH + ch];
  }
}

__global__ __launch_bounds__(256) void k_scan_p3(
    const float* __restrict__ Ag, float* __restrict__ Bg,
    const float* __restrict__ SC)
{
  int g  = blockIdx.x * 256 + threadIdx.x;
  int ch = g & (NCH - 1), c = g >> 13;
  int b = ch >> 10, hd = ch & 1023;
  size_t base = (size_t)b * 4096 * 1024 + (size_t)c * CHUNK * 1024 + hd;
  float h = SC[(size_t)c * NCH + ch];
  for (int t = 0; t < CHUNK; ++t) {
    size_t idx = base + (size_t)t * 1024;
    if (t + 4 < CHUNK) {
      __builtin_prefetch(Ag + idx + 4 * 1024, 0, 1);
      __builtin_prefetch(Bg + idx + 4 * 1024, 0, 1);
    }
    h = Ag[idx] * h + Bg[idx];
    Bg[idx] = h;
  }
}

// ---------------------------------------------------------------------------
// Per-token head mix: out_h = hs_h @ Wc_h + bc_h + dx; per-head LN;
// z = (dx + sum_h w_h * out_h)/16; u = z + LN(z).  One 64-thread block/token.
// ---------------------------------------------------------------------------
__global__ __launch_bounds__(64) void k_token_mix(
    const float* __restrict__ HS, const float* __restrict__ DX,
    const float* __restrict__ Wc, const float* __restrict__ bc,
    const float* __restrict__ head_w, const float* __restrict__ hn_g,
    const float* __restrict__ hn_b, const float* __restrict__ ng,
    const float* __restrict__ nb, float* __restrict__ U)
{
  __shared__ float hsS[1024];
  __shared__ float r1[64], r2[64];
  const int n = threadIdx.x;
  const size_t tok = blockIdx.x;
  #pragma unroll
  for (int i = 0; i < 16; ++i) hsS[n + i * 64] = HS[tok * 1024 + n + i * 64];
  __syncthreads();

  const float dxv = DX[tok * 64 + n];
  float zacc = 0.f;
  for (int h = 0; h < 16; ++h) {
    const float* hh = &hsS[h * 64];
    const float* w  = Wc + (size_t)h * 4096 + n;      // Wc[h][d][n], stride 64 in d
    float v = bc[h * 64 + n] + dxv;
    #pragma unroll 8
    for (int d = 0; d < 64; ++d) v += hh[d] * w[d * 64];
    r1[n] = v; r2[n] = v * v;
    __syncthreads();
    for (int s = 32; s > 0; s >>= 1) {
      if (n < s) { r1[n] += r1[n + s]; r2[n] += r2[n + s]; }
      __syncthreads();
    }
    float mean = r1[0] * (1.f / 64.f);
    float var  = r2[0] * (1.f / 64.f) - mean * mean;
    float o = (v - mean) * rsqrtf(var + 1e-5f) * hn_g[h * 64 + n] + hn_b[h * 64 + n];
    zacc += head_w[h] * o;
    __syncthreads();
  }
  float z = (dxv + zacc) * (1.f / 16.f);
  r1[n] = z; r2[n] = z * z;
  __syncthreads();
  for (int s = 32; s > 0; s >>= 1) {
    if (n < s) { r1[n] += r1[n + s]; r2[n] += r2[n + s]; }
    __syncthreads();
  }
  float mean = r1[0] * (1.f / 64.f);
  float var  = r2[0] * (1.f / 64.f) - mean * mean;
  float u = z + (z - mean) * rsqrtf(var + 1e-5f) * ng[n] + nb[n];
  U[tok * 64 + n] = u;
}

// ---------------------------------------------------------------------------
// FFN + output projection (WMMA): t1 = gelu(u@W1+b1); f = t1@W2+b2;
// h = u + f; out = h@Wp + bp.  4 waves/block, 16 tokens/wave, 64 tokens/block.
// ---------------------------------------------------------------------------
__global__ __launch_bounds__(128) void k_ffn(
    const float* __restrict__ U, const _Float16* __restrict__ W1t,
    const float* __restrict__ b1, const _Float16* __restrict__ W2t,
    const float* __restrict__ b2, const _Float16* __restrict__ Wpt,
    const float* __restrict__ bp, float* __restrict__ out)
{
  __shared__ __align__(16) _Float16 Us[64][64];
  __shared__ __align__(16) _Float16 T1[4][16][256];
  __shared__ __align__(16) _Float16 Hs2[4][16][64];
  const int tid = threadIdx.x;
  const int w = tid >> 5, lane = tid & 31, r = lane & 15, hf = lane >> 4;
  const int tok0 = blockIdx.x * 64;
  const v8f zz = {0.f,0.f,0.f,0.f,0.f,0.f,0.f,0.f};

  #pragma unroll
  for (int i = 0; i < 8; ++i) {
    int c = tid + i * 128, trow = c >> 4, pos = (c & 15) * 4;
    f32x4 v = *(const f32x4*)(U + (size_t)(tok0 + trow) * 64 + pos);
    _Float16* d = &Us[trow][pos];
    d[0] = (_Float16)v.x; d[1] = (_Float16)v.y; d[2] = (_Float16)v.z; d[3] = (_Float16)v.w;
  }
  __syncthreads();
  const int tw = w * 16;

  // GEMM1: [16,64] x [64,256] -> gelu -> T1
  FragH a1[2];
  #pragma unroll
  for (int ks = 0; ks < 2; ++ks) {
    const _Float16* rp = &Us[tw + r][0];
    a1[ks].q[0] = *(const f32x4*)(rp + ks * 32 + hf * 8);
    a1[ks].q[1] = *(const f32x4*)(rp + ks * 32 + 16 + hf * 8);
  }
  #pragma unroll 4
  for (int nt = 0; nt < 16; ++nt) {
    int col = nt * 16 + r;
    v8f z = zz;
    #pragma unroll
    for (int ks = 0; ks < 2; ++ks) {
      FragH b;
      const _Float16* wp = W1t + (size_t)col * 64 + ks * 32 + hf * 8;
      b.q[0] = *(const f32x4*)(wp);
      b.q[1] = *(const f32x4*)(wp + 16);
      z = __builtin_amdgcn_wmma_f32_16x16x32_f16(false, a1[ks].h, false, b.h,
                                                 (short)0, z, false, false);
    }
    float bias = b1[col];
    #pragma unroll
    for (int v = 0; v < 8; ++v) {
      float x = z[v] + bias;
      float g = 0.5f * x * (1.f + erff(x * 0.70710678118654752f));   // exact gelu
      T1[w][v + 8 * hf][col] = (_Float16)g;
    }
  }
  __syncthreads();

  // GEMM2: [16,256] x [256,64] + b2 + residual u -> Hs2
  v8f z2[4];
  #pragma unroll
  for (int nt = 0; nt < 4; ++nt) z2[nt] = zz;
  #pragma unroll
  for (int ks = 0; ks < 8; ++ks) {
    FragH a2;
    const _Float16* rp = &T1[w][r][0];
    a2.q[0] = *(const f32x4*)(rp + ks * 32 + hf * 8);
    a2.q[1] = *(const f32x4*)(rp + ks * 32 + 16 + hf * 8);
    #pragma unroll
    for (int nt = 0; nt < 4; ++nt) {
      int col = nt * 16 + r;
      FragH b;
      const _Float16* wp = W2t + (size_t)col * 256 + ks * 32 + hf * 8;
      b.q[0] = *(const f32x4*)(wp);
      b.q[1] = *(const f32x4*)(wp + 16);
      z2[nt] = __builtin_amdgcn_wmma_f32_16x16x32_f16(false, a2.h, false, b.h,
                                                      (short)0, z2[nt], false, false);
    }
  }
  #pragma unroll
  for (int nt = 0; nt < 4; ++nt) {
    int col = nt * 16 + r;
    float bias = b2[col];
    #pragma unroll
    for (int v = 0; v < 8; ++v) {
      int m = v + 8 * hf;
      float x = z2[nt][v] + bias + U[(size_t)(tok0 + tw + m) * 64 + col];
      Hs2[w][m][col] = (_Float16)x;
    }
  }
  __syncthreads();

  // GEMM3: [16,64] x [64,64] + bp -> out
  FragH a3[2];
  #pragma unroll
  for (int ks = 0; ks < 2; ++ks) {
    const _Float16* rp = &Hs2[w][r][0];
    a3[ks].q[0] = *(const f32x4*)(rp + ks * 32 + hf * 8);
    a3[ks].q[1] = *(const f32x4*)(rp + ks * 32 + 16 + hf * 8);
  }
  #pragma unroll
  for (int nt = 0; nt < 4; ++nt) {
    int col = nt * 16 + r;
    v8f z3 = zz;
    #pragma unroll
    for (int ks = 0; ks < 2; ++ks) {
      FragH b;
      const _Float16* wp = Wpt + (size_t)col * 64 + ks * 32 + hf * 8;
      b.q[0] = *(const f32x4*)(wp);
      b.q[1] = *(const f32x4*)(wp + 16);
      z3 = __builtin_amdgcn_wmma_f32_16x16x32_f16(false, a3[ks].h, false, b.h,
                                                  (short)0, z3, false, false);
    }
    float bias = bp[col];
    #pragma unroll
    for (int v = 0; v < 8; ++v) {
      int m = v + 8 * hf;
      out[(size_t)(tok0 + tw + m) * 64 + col] = z3[v] + bias;
    }
  }
}

// ---------------------------------------------------------------------------
extern "C" void kernel_launch(void* const* d_in, const int* in_sizes, int n_in,
                              void* d_out, int out_size, void* d_ws, size_t ws_size,
                              hipStream_t stream)
{
  (void)in_sizes; (void)n_in; (void)out_size; (void)ws_size;
  const float* emb    = (const float*)d_in[0];
  const float* Wa     = (const float*)d_in[1];
  const float* ba     = (const float*)d_in[2];
  const float* Wb     = (const float*)d_in[3];
  const float* bb     = (const float*)d_in[4];
  const float* Wc     = (const float*)d_in[5];
  const float* bc     = (const float*)d_in[6];
  const float* head_w = (const float*)d_in[7];
  const float* hn_g   = (const float*)d_in[8];
  const float* hn_b   = (const float*)d_in[9];
  const float* Wd     = (const float*)d_in[10];
  const float* bd     = (const float*)d_in[11];
  const float* W1     = (const float*)d_in[12];
  const float* b1     = (const float*)d_in[13];
  const float* W2     = (const float*)d_in[14];
  const float* b2     = (const float*)d_in[15];
  const float* Wp     = (const float*)d_in[16];
  const float* bp     = (const float*)d_in[17];
  const float* ng     = (const float*)d_in[18];
  const float* nb     = (const float*)d_in[19];
  float* out = (float*)d_out;

  auto al = [](size_t x) { return (x + 255) & ~(size_t)255; };
  char* p = (char*)d_ws;
  _Float16* wcat = (_Float16*)p; p += al((size_t)N_PAD * K_E * 2);
  _Float16* w1t  = (_Float16*)p; p += al((size_t)256 * 64 * 2);
  _Float16* w2t  = (_Float16*)p; p += al((size_t)64 * 256 * 2);
  _Float16* wpt  = (_Float16*)p; p += al((size_t)64 * 64 * 2);
  float* wsA  = (float*)p; p += al((size_t)M_TOT * 1024 * 4);   // a gates
  float* wsB  = (float*)p; p += al((size_t)M_TOT * 1024 * 4);   // b gates -> hs
  float* wsDX = (float*)p; p += al((size_t)M_TOT * 64 * 4);     // dx
  float* wsU  = (float*)p; p += al((size_t)M_TOT * 64 * 4);     // u
  float* wsSA = (float*)p; p += al((size_t)NCH * NCHK * 4);     // chunk prod(a)
  float* wsSB = (float*)p; p += al((size_t)NCH * NCHK * 4);     // chunk scan-from-0
  float* wsSC = (float*)p; p += al((size_t)NCH * NCHK * 4);     // chunk carries

  k_pack_wcat<<<(N_PAD * K_E) / 256, 256, 0, stream>>>(Wa, Wb, Wd, wcat);
  k_pack_trans<<<(64 * 256) / 256, 256, 0, stream>>>(W1, w1t, 64, 256);
  k_pack_trans<<<(256 * 64) / 256, 256, 0, stream>>>(W2, w2t, 256, 64);
  k_pack_trans<<<(64 * 64) / 256, 256, 0, stream>>>(Wp, wpt, 64, 64);

  k_gemm_proj<<<dim3(N_PAD / 128, M_TOT / 128), 256, 0, stream>>>(
      emb, wcat, ba, bb, bd, wsA, wsB, wsDX);

  k_scan_p1<<<(NCH * NCHK) / 256, 256, 0, stream>>>(wsA, wsB, wsSA, wsSB);
  k_scan_p2<<<NCH / 256, 256, 0, stream>>>(wsSA, wsSB, wsSC);
  k_scan_p3<<<(NCH * NCHK) / 256, 256, 0, stream>>>(wsA, wsB, wsSC);

  k_token_mix<<<M_TOT, 64, 0, stream>>>(wsB, wsDX, Wc, bc, head_w, hn_g, hn_b,
                                        ng, nb, wsU);

  k_ffn<<<M_TOT / 64, 128, 0, stream>>>(wsU, w1t, b1, w2t, b2, wpt, bp, out);
}
